// Decoder_86157043958297
// MI455X (gfx1250) — matrix operand order repaired and checked by reading, not compiled, not run on hardware
//
#include <hip/hip_runtime.h>
#include <math.h>

// ---------------------------------------------------------------------------
// Problem constants (from the reference):
//   B=64, S=128, E=512, H=1024, A=1024, M=512, V=32000
// ---------------------------------------------------------------------------
#define CB 64
#define CS 128
#define CE 512
#define CH 1024
#define CA 1024
#define CM 512
#define CV 32000

typedef __attribute__((ext_vector_type(16))) __bf16 v16bf;
typedef __attribute__((ext_vector_type(8)))  float  v8f;
typedef __attribute__((ext_vector_type(4)))  unsigned int uint4v;
typedef __attribute__((ext_vector_type(8)))  int    int8v;
typedef __attribute__((ext_vector_type(4)))  int    int4v;

union Frag {
    v16bf  v;
    __bf16 h[16];
};

// This toolchain (clang-23 / therock) exposes the 6-arg TDM builtin:
//   (uint32x4 g0, int32x8 g1, int32x4 g2, int32x4 g3, int32x8, i32 cpol)
#ifndef USE_TDM
#  if defined(__has_builtin)
#    if __has_builtin(__builtin_amdgcn_tensor_load_to_lds) && \
        __has_builtin(__builtin_amdgcn_s_wait_tensorcnt)
#      define USE_TDM 1
#    else
#      define USE_TDM 0
#    endif
#  else
#    define USE_TDM 0
#  endif
#endif

// ---------------------------------------------------------------------------
// GEMM tiling: C(M,N) = A(M,K) @ W(N,K)^T + bias(N)
// Block tile 64(M) x 128(N) x 32(K), 256 threads = 8 waves.
// Wave (wm=w&1, wn=w>>1) owns a 32x32 output slab: 2 A-frags x 2 B-frags,
// 4 x v_wmma_f32_16x16x32_bf16 per K-step, f32 accumulation.
// LDS tiles are raw f32 (TDM-copied), padded to stride 36 dwords
// (TDM pad: 4 dwords every 32) to spread the 64 LDS banks.
// f32 -> bf16 RNE conversion happens in-register during fragment assembly
// (VALU cvt co-executes with the matrix pipe).
// ---------------------------------------------------------------------------
#define BM 64
#define BN 128
#define BK 32
#define LDF 36   // f32 elements per LDS row (32 data + 4 pad)

__device__ __forceinline__ unsigned lds_off(const void* p) {
    // AMDGPU generic pointers to LDS carry the LDS byte offset in the low
    // 32 bits (ISA 10.2).
    return (unsigned)(size_t)p;
}

#if USE_TDM
// Issue one TENSOR_LOAD_TO_LDS for a (rows x 32) f32 tile whose rows are
// k_stride f32 apart in memory.  LDS padding: 4 dwords every 32 dwords
// (pad_interval code 4 = 32 dwords, pad_amount code 3 = 4 dwords).
__device__ __forceinline__ void tdm_load_tile_f32(
    const float* gsrc, unsigned ldsbyte, int k_stride, int rows)
{
    unsigned long long ga = (unsigned long long)(size_t)gsrc;
    uint4v g0;
    g0[0] = 1u;                                        // count=1, user mode
    g0[1] = ldsbyte;                                   // lds_addr
    g0[2] = (unsigned)ga;                              // global_addr[31:0]
    g0[3] = ((unsigned)(ga >> 32) & 0x1FFFFFFu)        // global_addr[56:32]
          | (2u << 30);                                // type = 2 ("image")

    int8v g1;
    g1[0] = (2 << 16)      // data_size = 4 bytes
          | (1 << 20)      // pad_enable
          | (4 << 22)      // pad_interval: 32 dwords
          | (3 << 25);     // pad_amount: 4 dwords
    g1[1] = (int)0xFFFF0000u;                 // tensor_dim0 = 0x7FFFFFFF (lo16<<16)
    g1[2] = (int)0xFFFF7FFFu;                 // tensor_dim0 hi16 | tensor_dim1 lo16
    g1[3] = (int)((BK << 16) | 0x7FFF);       // tensor_dim1 hi16 | tile_dim0 = 32
    g1[4] = rows;                             // tile_dim1 = rows, tile_dim2 = 0
    g1[5] = k_stride;                         // tensor_dim0_stride[31:0]
    g1[6] = 0;                                // stride hi / dim1_stride lo
    g1[7] = 0;

    int4v z4 = {0, 0, 0, 0};
    int8v z8 = {0, 0, 0, 0, 0, 0, 0, 0};
    __builtin_amdgcn_tensor_load_to_lds(g0, g1, z4, z4, z8, 0);
}
#endif

// A fragment: 16x32 bf16 (ISA 7.12.2). Lane(l16, half): row r, two 32B runs
// of f32 at k = 8*half..8*half+7 and 16+8*half..+7, converted to bf16.
__device__ __forceinline__ Frag load_afrag(const float* __restrict__ as,
                                           int r, int half)
{
    const float* p = as + r * LDF;
    float4 x0 = *(const float4*)(p + 8 * half);
    float4 x1 = *(const float4*)(p + 8 * half + 4);
    float4 x2 = *(const float4*)(p + 16 + 8 * half);
    float4 x3 = *(const float4*)(p + 16 + 8 * half + 4);
    Frag f;
    f.h[0]  = (__bf16)x0.x; f.h[1]  = (__bf16)x0.y;
    f.h[2]  = (__bf16)x0.z; f.h[3]  = (__bf16)x0.w;
    f.h[4]  = (__bf16)x1.x; f.h[5]  = (__bf16)x1.y;
    f.h[6]  = (__bf16)x1.z; f.h[7]  = (__bf16)x1.w;
    f.h[8]  = (__bf16)x2.x; f.h[9]  = (__bf16)x2.y;
    f.h[10] = (__bf16)x2.z; f.h[11] = (__bf16)x2.w;
    f.h[12] = (__bf16)x3.x; f.h[13] = (__bf16)x3.y;
    f.h[14] = (__bf16)x3.z; f.h[15] = (__bf16)x3.w;
    return f;
}

// B fragment: 32x16 bf16. Lane(l16, half): column row r, two 32B runs of
// f32 at k = 8*half..8*half+7 and 16+8*half..+7.
__device__ __forceinline__ Frag load_bfrag(const float* __restrict__ bs,
                                           int r, int half)
{
    const float* p = bs + r * LDF;
    float4 x0 = *(const float4*)(p + 8 * half);
    float4 x1 = *(const float4*)(p + 8 * half + 4);
    float4 x2 = *(const float4*)(p + 16 + 8 * half);
    float4 x3 = *(const float4*)(p + 16 + 8 * half + 4);
    Frag f;
    f.h[0]  = (__bf16)x0.x; f.h[1]  = (__bf16)x0.y;
    f.h[2]  = (__bf16)x0.z; f.h[3]  = (__bf16)x0.w;
    f.h[4]  = (__bf16)x1.x; f.h[5]  = (__bf16)x1.y;
    f.h[6]  = (__bf16)x1.z; f.h[7]  = (__bf16)x1.w;
    f.h[8]  = (__bf16)x2.x; f.h[9]  = (__bf16)x2.y;
    f.h[10] = (__bf16)x2.z; f.h[11] = (__bf16)x2.w;
    f.h[12] = (__bf16)x3.x; f.h[13] = (__bf16)x3.y;
    f.h[14] = (__bf16)x3.z; f.h[15] = (__bf16)x3.w;
    return f;
}

__device__ __forceinline__ void mma_step(
    const float* __restrict__ as, const float* __restrict__ bs,
    int wm, int wn, int half, int l16,
    v8f& acc00, v8f& acc01, v8f& acc10, v8f& acc11)
{
    Frag a0 = load_afrag(as, 32 * wm + l16, half);
    Frag a1 = load_afrag(as, 32 * wm + 16 + l16, half);
    Frag b0 = load_bfrag(bs, 32 * wn + l16, half);
    Frag b1 = load_bfrag(bs, 32 * wn + 16 + l16, half);
    acc00 = __builtin_amdgcn_wmma_f32_16x16x32_bf16(
        false, a0.v, false, b0.v, (short)0, acc00, false, false);
    acc01 = __builtin_amdgcn_wmma_f32_16x16x32_bf16(
        false, a0.v, false, b1.v, (short)0, acc01, false, false);
    acc10 = __builtin_amdgcn_wmma_f32_16x16x32_bf16(
        false, a1.v, false, b0.v, (short)0, acc10, false, false);
    acc11 = __builtin_amdgcn_wmma_f32_16x16x32_bf16(
        false, a1.v, false, b1.v, (short)0, acc11, false, false);
}

__global__ __launch_bounds__(256) void gemm_bf16_wmma(
    const float* __restrict__ A,     // (M,K) row-major
    const float* __restrict__ W,     // (N,K) row-major
    const float* __restrict__ bias,  // (N)
    float* __restrict__ C,           // (M,N) row-major
    int M, int N, int K)
{
    const int tid  = threadIdx.x;
    const int lane = tid & 31;
    const int wave = tid >> 5;       // 0..7
    const int wm   = wave & 1;       // 32-row half of the 64-row tile
    const int wn   = wave >> 1;      // 32-col quarter of the 128-col tile
    const int half = lane >> 4;
    const int l16  = lane & 15;
    const int m0   = blockIdx.y * BM;
    const int n0   = blockIdx.x * BN;
    const int nk   = K / BK;

    v8f acc00 = {}, acc01 = {}, acc10 = {}, acc11 = {};

#if USE_TDM
    __shared__ __align__(16) float As[2][BM * LDF];
    __shared__ __align__(16) float Bs[2][BN * LDF];

    // Prime the pipeline: DMA tile 0 into buffer 0 (wave 0 issues; TENSORcnt
    // is per-wave, the barrier below publishes completion to the block).
    if (wave == 0) {
        tdm_load_tile_f32(A + (size_t)m0 * K, lds_off(&As[0][0]), K, BM);
        tdm_load_tile_f32(W + (size_t)n0 * K, lds_off(&Bs[0][0]), K, BN);
    }
    for (int t = 0; t < nk; ++t) {
        const int cur = t & 1;
        if (wave == 0) __builtin_amdgcn_s_wait_tensorcnt(0);
        __syncthreads();   // tile t visible to all waves; tile t-1 reads done
        if (wave == 0 && (t + 1) < nk) {
            const int k1 = (t + 1) * BK;
            tdm_load_tile_f32(A + (size_t)m0 * K + k1,
                              lds_off(&As[cur ^ 1][0]), K, BM);
            tdm_load_tile_f32(W + (size_t)n0 * K + k1,
                              lds_off(&Bs[cur ^ 1][0]), K, BN);
        }
        mma_step(As[cur], Bs[cur], wm, wn, half, l16,
                 acc00, acc01, acc10, acc11);
    }
#else
    __shared__ __align__(16) float As[1][BM * LDF];
    __shared__ __align__(16) float Bs[1][BN * LDF];

    for (int t = 0; t < nk; ++t) {
        const int k0 = t * BK;
        #pragma unroll
        for (int i = 0; i < (BM * BK) / (256 * 4); ++i) {   // 2 float4 / thread
            int q = tid + i * 256;
            int r = q >> 3, c = (q & 7) * 4;
            *(float4*)&As[0][r * LDF + c] =
                *(const float4*)&A[(size_t)(m0 + r) * K + k0 + c];
        }
        #pragma unroll
        for (int i = 0; i < (BN * BK) / (256 * 4); ++i) {   // 4 float4 / thread
            int q = tid + i * 256;
            int r = q >> 3, c = (q & 7) * 4;
            *(float4*)&Bs[0][r * LDF + c] =
                *(const float4*)&W[(size_t)(n0 + r) * K + k0 + c];
        }
        __syncthreads();
        mma_step(As[0], Bs[0], wm, wn, half, l16,
                 acc00, acc01, acc10, acc11);
        __syncthreads();
    }
#endif

    // Epilogue: C fragment layout (VGPR r -> rows r / r+8 per lane half).
    #pragma unroll
    for (int r = 0; r < 8; ++r) {
        int mA = m0 + 32 * wm + r + 8 * half;
        int mB = mA + 16;
        int nA = n0 + 32 * wn + l16;
        int nB = nA + 16;
        C[(size_t)mA * N + nA] = acc00[r] + bias[nA];
        C[(size_t)mA * N + nB] = acc01[r] + bias[nB];
        C[(size_t)mB * N + nA] = acc10[r] + bias[nA];
        C[(size_t)mB * N + nB] = acc11[r] + bias[nB];
    }
}

// ---------------------------------------------------------------------------
// Embedding gather: x[:, :E] = y[:, :E] = embed_table[inputs]
// ---------------------------------------------------------------------------
__global__ __launch_bounds__(256) void gather_embed(
    const int* __restrict__ inputs, const float* __restrict__ table,
    float* __restrict__ xbuf, float* __restrict__ ybuf)
{
    int idx = blockIdx.x * 256 + threadIdx.x;     // B*E = 32768
    int b = idx >> 9;
    int e = idx & (CE - 1);
    float val = table[(size_t)inputs[b] * CE + e];
    xbuf[b * (CE + 2 * CH) + e] = val;
    ybuf[b * (CE + 2 * CH + CH) + e] = val;
}

// ---------------------------------------------------------------------------
// Attention: one block per batch element.
// ---------------------------------------------------------------------------
__global__ __launch_bounds__(256) void attn_kernel(
    const float* __restrict__ enc_hs,   // (B*S, A)
    const float* __restrict__ dec_h,    // (B, A)
    const float* __restrict__ vvec,     // (A)
    const float* __restrict__ enc_raw,  // (B, S, 2H)
    float* __restrict__ xbuf,           // (B, E+2H)
    float* __restrict__ ybuf)           // (B, E+2H+H)
{
    const int b    = blockIdx.x;
    const int tid  = threadIdx.x;
    const int lane = tid & 31;
    const int wave = tid >> 5;

    __shared__ float s_dec[CA];
    __shared__ float s_v[CA];
    __shared__ float s_logit[CS];
    __shared__ int   s_nz[CS];
    __shared__ float s_score[CS];

    for (int i = tid; i < CA; i += 256) {
        s_dec[i] = dec_h[b * CA + i];
        s_v[i]   = vvec[i];
    }
    __syncthreads();

    for (int s = wave; s < CS; s += 8) {
        const float* row = enc_hs + ((size_t)(b * CS + s)) * CA;
        float acc = 0.f;
        int   nz  = 0;
        for (int a = lane; a < CA; a += 32) {
            float e = row[a];
            nz |= (e != 0.0f) ? 1 : 0;
            acc += tanhf(e + s_dec[a]) * s_v[a];
        }
        #pragma unroll
        for (int off = 16; off > 0; off >>= 1) {
            acc += __shfl_down(acc, off, 32);
            nz  |= __shfl_down(nz,  off, 32);
        }
        if (lane == 0) { s_logit[s] = acc; s_nz[s] = nz; }
    }
    __syncthreads();

    if (tid == 0) {
        int valid = 1;
        float mx = -INFINITY;
        for (int s = 0; s < CS; ++s) {
            valid = valid && s_nz[s];
            s_nz[s] = valid;
            if (valid && s_logit[s] > mx) mx = s_logit[s];
        }
        float denom = 0.f;
        for (int s = 0; s < CS; ++s) {
            float e = s_nz[s] ? expf(s_logit[s] - mx) : 0.f;
            s_score[s] = e;
            denom += e;
        }
        float inv = (denom > 0.f) ? 1.f / denom : 0.f;
        for (int s = 0; s < CS; ++s) s_score[s] *= inv;
    }
    __syncthreads();

    for (int c = tid; c < 2 * CH; c += 256) {
        const float* base = enc_raw + ((size_t)b * CS) * (2 * CH) + c;
        float acc = 0.f;
        for (int s = 0; s < CS; ++s) acc += s_score[s] * base[(size_t)s * (2 * CH)];
        xbuf[b * (CE + 2 * CH) + CE + c]      = acc;
        ybuf[b * (CE + 2 * CH + CH) + CE + c] = acc;
    }
}

// ---------------------------------------------------------------------------
// GRU gate fusion + new_h writeout (into y buffer and d_out tail)
// ---------------------------------------------------------------------------
__global__ __launch_bounds__(256) void gru_kernel(
    const float* __restrict__ gi, const float* __restrict__ gh,
    const float* __restrict__ hidden,
    float* __restrict__ ybuf, float* __restrict__ out_h)
{
    int idx = blockIdx.x * 256 + threadIdx.x;  // B*H = 65536
    int b = idx >> 10;
    int j = idx & (CH - 1);
    float ir = gi[b * 3 * CH + j];
    float iz = gi[b * 3 * CH + CH + j];
    float in = gi[b * 3 * CH + 2 * CH + j];
    float hr = gh[b * 3 * CH + j];
    float hz = gh[b * 3 * CH + CH + j];
    float hn = gh[b * 3 * CH + 2 * CH + j];
    float r = 1.f / (1.f + expf(-(ir + hr)));
    float z = 1.f / (1.f + expf(-(iz + hz)));
    float n = tanhf(in + r * hn);
    float h = hidden[idx];
    float nh = (1.f - z) * n + z * h;
    ybuf[b * (CE + 2 * CH + CH) + (CE + 2 * CH) + j] = nh;
    out_h[idx] = nh;
}

// ---------------------------------------------------------------------------
// Maxout pair-reduction: m[b,j] = max(mm[b,2j], mm[b,2j+1])
// ---------------------------------------------------------------------------
__global__ __launch_bounds__(256) void maxout_reduce(
    const float* __restrict__ mm, float* __restrict__ m)
{
    int idx = blockIdx.x * 256 + threadIdx.x;  // B*M = 32768
    int b = idx >> 9;
    int j = idx & (CM - 1);
    m[idx] = fmaxf(mm[b * 2 * CM + 2 * j], mm[b * 2 * CM + 2 * j + 1]);
}

// ---------------------------------------------------------------------------
// Launch
// ---------------------------------------------------------------------------
extern "C" void kernel_launch(void* const* d_in, const int* in_sizes, int n_in,
                              void* d_out, int out_size, void* d_ws, size_t ws_size,
                              hipStream_t stream) {
    (void)in_sizes; (void)n_in; (void)out_size; (void)ws_size;

    const int*   inputs          = (const int*)  d_in[0];
    const float* hidden          = (const float*)d_in[1];   // (B,1,H)
    const float* encoder_hiddens = (const float*)d_in[2];   // (B,S,2H)
    const float* embed_table     = (const float*)d_in[3];   // (V,E)
    const float* enc_W           = (const float*)d_in[4];   // (A,2H)
    const float* enc_b           = (const float*)d_in[5];
    const float* dec_W           = (const float*)d_in[6];   // (A,H)
    const float* dec_b           = (const float*)d_in[7];
    const float* vvec            = (const float*)d_in[8];   // (1,A)
    const float* W_ih            = (const float*)d_in[9];   // (3H, E+2H)
    const float* b_ih            = (const float*)d_in[10];
    const float* W_hh            = (const float*)d_in[11];  // (3H, H)
    const float* b_hh            = (const float*)d_in[12];
    const float* maxout_W        = (const float*)d_in[13];  // (2M, 3H+E)
    const float* maxout_b        = (const float*)d_in[14];
    const float* out_W           = (const float*)d_in[15];  // (V, M)
    const float* out_b           = (const float*)d_in[16];

    float* out   = (float*)d_out;                 // [B*V outputs][B*H new_h]
    float* out_h = out + (size_t)CB * CV;

    float* ws     = (float*)d_ws;
    float* enc_hs = ws;                                        // B*S x A
    float* dec_h  = enc_hs + (size_t)CB * CS * CA;             // B x A
    float* xbuf   = dec_h  + (size_t)CB * CA;                  // B x (E+2H)
    float* ybuf   = xbuf   + (size_t)CB * (CE + 2 * CH);       // B x (E+2H+H)
    float* gi     = ybuf   + (size_t)CB * (CE + 2 * CH + CH);  // B x 3H
    float* gh     = gi     + (size_t)CB * 3 * CH;              // B x 3H
    float* mm     = gh     + (size_t)CB * 3 * CH;              // B x 2M
    float* mred   = mm     + (size_t)CB * 2 * CM;              // B x M

    dim3 blk(256);

    // 1) enc_hs = encoder_hiddens @ enc_W^T + enc_b   (8192 x 1024 x 2048)
    gemm_bf16_wmma<<<dim3(CA / BN, (CB * CS) / BM), blk, 0, stream>>>(
        encoder_hiddens, enc_W, enc_b, enc_hs, CB * CS, CA, 2 * CH);

    // 2) dec_h = h @ dec_W^T + dec_b                  (64 x 1024 x 1024)
    gemm_bf16_wmma<<<dim3(CA / BN, CB / BM), blk, 0, stream>>>(
        hidden, dec_W, dec_b, dec_h, CB, CA, CH);

    // 3) embedding gather
    gather_embed<<<dim3((CB * CE) / 256), blk, 0, stream>>>(
        inputs, embed_table, xbuf, ybuf);

    // 4) attention: logits/softmax/context
    attn_kernel<<<dim3(CB), blk, 0, stream>>>(
        enc_hs, dec_h, vvec, encoder_hiddens, xbuf, ybuf);

    // 5) gi = [embed, context] @ W_ih^T + b_ih        (64 x 3072 x 2560)
    gemm_bf16_wmma<<<dim3((3 * CH) / BN, CB / BM), blk, 0, stream>>>(
        xbuf, W_ih, b_ih, gi, CB, 3 * CH, CE + 2 * CH);

    // 6) gh = h @ W_hh^T + b_hh                       (64 x 3072 x 1024)
    gemm_bf16_wmma<<<dim3((3 * CH) / BN, CB / BM), blk, 0, stream>>>(
        hidden, W_hh, b_hh, gh, CB, 3 * CH, CH);

    // 7) GRU gates -> new_h
    gru_kernel<<<dim3((CB * CH) / 256), blk, 0, stream>>>(
        gi, gh, hidden, ybuf, out_h);

    // 8) mm = [embed, context, new_h] @ maxout_W^T + maxout_b (64x1024x3584)
    gemm_bf16_wmma<<<dim3((2 * CM) / BN, CB / BM), blk, 0, stream>>>(
        ybuf, maxout_W, maxout_b, mm, CB, 2 * CM, CE + 2 * CH + CH);

    // 9) maxout pair-max
    maxout_reduce<<<dim3((CB * CM) / 256), blk, 0, stream>>>(mm, mred);

    // 10) outputs = m @ out_W^T + out_b               (64 x 32000 x 512)
    gemm_bf16_wmma<<<dim3(CV / BN, CB / BM), blk, 0, stream>>>(
        mred, out_W, out_b, out, CB, CV, CM);
}
